// AttentionSublayer_10479720202996
// MI455X (gfx1250) — matrix operand order repaired and checked by reading, not compiled, not run on hardware
//
#include <hip/hip_runtime.h>
#include <cstdint>

// ---------------------------------------------------------------------------
// MI455X / gfx1250 attention sublayer with relative position embeddings.
// All matmuls via v_wmma_f32_16x16x32_bf16 (bf16 operands, f32 accumulate).
// fp32 -> bf16 conversion is a separate streaming pass so the GEMM hot loops
// do pure b128 loads into distinct fragment registers (no load->wait->cvt
// serialization).
// ---------------------------------------------------------------------------

typedef __bf16 bf16;
typedef __bf16 bf16x16 __attribute__((ext_vector_type(16)));
typedef __bf16 bf16x8  __attribute__((ext_vector_type(8)));
typedef float  f32x8   __attribute__((ext_vector_type(8)));

#define DEV __device__ __forceinline__

static constexpr int Bn = 4;     // batch
static constexpr int Hn = 16;    // heads
static constexpr int Tn = 1024;  // seq len
static constexpr int Dn = 1024;  // model dim
static constexpr int DH = 64;    // head dim
static constexpr int Ln = 128;   // max rel pos
static constexpr int RELW = 272; // 257 padded to 16-tile multiple

DEV f32x8 zero8() {
  f32x8 z;
#pragma unroll
  for (int i = 0; i < 8; ++i) z[i] = 0.0f;
  return z;
}

DEV f32x8 wmma_bf16(bf16x16 a, bf16x16 b, f32x8 c) {
  // D = A(16x32) * B(32x16) + C, f32 accumulate
  return __builtin_amdgcn_wmma_f32_16x16x32_bf16(
      /*neg_a=*/false, a, /*neg_b=*/false, b,
      /*c_mod=*/(short)0, c, /*reuse_a=*/false, /*reuse_b=*/false);
}

// A-fragment (16x32, 16-bit): lane r=lane&15 holds row r.
// f[0..15] = row[hi*16 + 0..15]   (contiguous K, matches B-fragment)
DEV bf16x16 ld_a_bf16(const bf16* row, int hi) {
  bf16x8 lo = *(const bf16x8*)(row + hi * 16);
  bf16x8 hb = *(const bf16x8*)(row + hi * 16 + 8);
  bf16x16 f;
#pragma unroll
  for (int i = 0; i < 8; ++i) { f[i] = lo[i]; f[8 + i] = hb[i]; }
  return f;
}

// B-fragment (32x16, 16-bit): lane r holds column r; 16 contiguous K at hi*16.
// Caller passes pointer to colsrc_row[n][k0 + hi*16] of the K-major source.
DEV bf16x16 ld_b_bf16(const bf16* col16) {
  bf16x8 a = *(const bf16x8*)col16;
  bf16x8 b = *(const bf16x8*)(col16 + 8);
  bf16x16 f;
#pragma unroll
  for (int i = 0; i < 8; ++i) { f[i] = a[i]; f[8 + i] = b[i]; }
  return f;
}

// ---------------------------------------------------------------------------
// Streaming fp32 -> bf16 conversion, 8 elements per thread, b128 in / b128 out.
// ---------------------------------------------------------------------------
__global__ void __launch_bounds__(256) cvt_bf16_kernel(
    const float* __restrict__ src, bf16* __restrict__ dst, unsigned n) {
  const unsigned idx = ((unsigned)blockIdx.x * 256u + threadIdx.x) * 8u;
  if (idx >= n) return;
  float4 a = *(const float4*)(src + idx);
  float4 b = *(const float4*)(src + idx + 4);
  bf16x8 o;
  o[0] = (bf16)a.x; o[1] = (bf16)a.y; o[2] = (bf16)a.z; o[3] = (bf16)a.w;
  o[4] = (bf16)b.x; o[5] = (bf16)b.y; o[6] = (bf16)b.z; o[7] = (bf16)b.w;
  *(bf16x8*)(dst + idx) = o;
}

// ---------------------------------------------------------------------------
// Projection GEMM: Y = X @ W^T.  X:(4096,1024) bf16, W:(1024,1024) bf16.
// Output bf16, mode 0: (B,H,T,DH) head-split (Q/K),
//              mode 1: (B,H,DH,T) head-split transposed (V).
// Block = 8 waves: 4 m-tiles x 2 n-groups -> 64x128 block tile.
// ---------------------------------------------------------------------------
__global__ void __launch_bounds__(256) gemm_proj_kernel(
    const bf16* __restrict__ X, const bf16* __restrict__ W,
    bf16* __restrict__ out, float scale, int mode) {
  const int lane = threadIdx.x & 31;
  const int r = lane & 15, hi = lane >> 4;
  const int wid = threadIdx.x >> 5;
  const int m0 = blockIdx.x * 64 + (wid & 3) * 16;
  const int n0 = blockIdx.y * 128 + (wid >> 2) * 64;

  f32x8 acc[4];
#pragma unroll
  for (int nt = 0; nt < 4; ++nt) acc[nt] = zero8();

  const bf16* arow = X + (size_t)(m0 + r) * Dn;
  const bf16* brow0 = W + (size_t)(n0 + r) * Dn;
  for (int k0 = 0; k0 < Dn; k0 += 32) {
    if (k0 + 256 < Dn) __builtin_prefetch(arow + k0 + 256, 0, 1);
    bf16x16 a = ld_a_bf16(arow + k0, hi);
    bf16x16 b0 = ld_b_bf16(brow0 + (size_t)0 * 16 * Dn + k0 + hi * 16);
    bf16x16 b1 = ld_b_bf16(brow0 + (size_t)1 * 16 * Dn + k0 + hi * 16);
    bf16x16 b2 = ld_b_bf16(brow0 + (size_t)2 * 16 * Dn + k0 + hi * 16);
    bf16x16 b3 = ld_b_bf16(brow0 + (size_t)3 * 16 * Dn + k0 + hi * 16);
    acc[0] = wmma_bf16(a, b0, acc[0]);
    acc[1] = wmma_bf16(a, b1, acc[1]);
    acc[2] = wmma_bf16(a, b2, acc[2]);
    acc[3] = wmma_bf16(a, b3, acc[3]);
  }
#pragma unroll
  for (int nt = 0; nt < 4; ++nt) {
    const int n = n0 + nt * 16 + r;
    const int h = n >> 6, dh = n & 63;
#pragma unroll
    for (int j = 0; j < 8; ++j) {
      const int m = m0 + j + 8 * hi;
      const int b = m >> 10, t = m & (Tn - 1);
      const bf16 v = (bf16)(acc[nt][j] * scale);
      if (mode == 0)
        out[((size_t)(b * Hn + h) * Tn + t) * DH + dh] = v;
      else
        out[((size_t)(b * Hn + h) * DH + dh) * Tn + t] = v;
    }
  }
}

// ---------------------------------------------------------------------------
// rel[m, i] = Qbf[m,:] . pos_emb[i,:]   m = (b,h,t) flat, i in [0,257)
// Qbf already scaled by 1/SCALE. Output bf16 (M, 272). Cols >=257 are junk
// (never gathered: clamp index max is 256).
// ---------------------------------------------------------------------------
__global__ void __launch_bounds__(256) relscore_kernel(
    const bf16* __restrict__ Qbf, const bf16* __restrict__ P,
    bf16* __restrict__ rel) {
  const int lane = threadIdx.x & 31;
  const int r = lane & 15, hi = lane >> 4;
  const int wid = threadIdx.x >> 5;
  const int m0 = blockIdx.x * 128 + wid * 16;

  const bf16* arow = Qbf + (size_t)(m0 + r) * DH;
  const bf16x16 a0 = ld_a_bf16(arow, hi);
  const bf16x16 a1 = ld_a_bf16(arow + 32, hi);

  for (int ntile = 0; ntile < RELW / 16; ++ntile) {
    const int n0 = ntile * 16;
    const int prow = (n0 + r > 256) ? 256 : (n0 + r);
    const bf16* col = P + (size_t)prow * DH;
    bf16x16 b0 = ld_b_bf16(col + hi * 16);
    bf16x16 b1 = ld_b_bf16(col + 32 + hi * 16);
    f32x8 acc = zero8();
    acc = wmma_bf16(a0, b0, acc);
    acc = wmma_bf16(a1, b1, acc);
#pragma unroll
    for (int j = 0; j < 8; ++j)
      rel[(size_t)(m0 + j + 8 * hi) * RELW + n0 + r] = (bf16)acc[j];
  }
}

// ---------------------------------------------------------------------------
// Flash-style attention per (b,h). Block = 8 waves x 16 q-rows = 128 q/block.
// Online softmax over key chunks of 32; K/V chunks staged in LDS, P tile
// transposed C-layout -> A-layout through per-wave LDS.
// ---------------------------------------------------------------------------
#define KPAD 72  // 64 + 8  (16B-aligned rows, conflict-spread)
#define VPAD 40  // 32 + 8

__global__ void __launch_bounds__(256) attn_kernel(
    const bf16* __restrict__ Qbf,   // (B,H,T,DH), pre-scaled 1/8
    const bf16* __restrict__ Kbf,   // (B,H,T,DH)
    const bf16* __restrict__ Vt,    // (B,H,DH,T)
    const bf16* __restrict__ rel,   // (B,H,T,RELW)
    const unsigned char* __restrict__ mask,  // (B,T) bool
    bf16* __restrict__ comb) {      // (B,T,D) heads concatenated
  __shared__ __align__(16) bf16 ldsK[32][KPAD];       // [t_local][dh]
  __shared__ __align__(16) bf16 ldsV[DH][VPAD];       // [dh][t_local]
  __shared__ __align__(16) bf16 ldsP[8][16][VPAD];    // per-wave [q][t_local]

  const int tid = threadIdx.x;
  const int lane = tid & 31, r = lane & 15, hi = lane >> 4;
  const int wid = tid >> 5;
  const int bh = blockIdx.y;           // b*H + h
  const int b = bh >> 4, h = bh & 15;
  const int q0 = blockIdx.x * 128 + wid * 16;

  const bf16* qrow = Qbf + ((size_t)bh * Tn + q0 + r) * DH;
  const bf16x16 qa0 = ld_a_bf16(qrow, hi);
  const bf16x16 qa1 = ld_a_bf16(qrow + 32, hi);

  f32x8 acc[4];
#pragma unroll
  for (int nt = 0; nt < 4; ++nt) acc[nt] = zero8();
  float mi[8], li[8];
#pragma unroll
  for (int j = 0; j < 8; ++j) { mi[j] = -1e30f; li[j] = 0.0f; }

  const bf16* Kbase = Kbf + (size_t)bh * Tn * DH;
  const bf16* Vbase = Vt + (size_t)bh * DH * Tn;
  const bf16* relbase = rel + (size_t)bh * Tn * RELW;
  const unsigned char* mrow = mask + (size_t)b * Tn;

  for (int tc = 0; tc < Tn; tc += 32) {
    __syncthreads();  // protect previous iteration's LDS consumers
    {
      const int kr = tid >> 3, ks = tid & 7;  // 32 rows x 128B
      *(uint4*)&ldsK[kr][ks * 8] =
          *(const uint4*)(Kbase + (size_t)(tc + kr) * DH + ks * 8);
      const int vr = tid >> 2, vs = tid & 3;  // 64 rows x 64B
      *(uint4*)&ldsV[vr][vs * 8] =
          *(const uint4*)(Vbase + (size_t)vr * Tn + tc + vs * 8);
    }
    __syncthreads();

    // ---- S = Q K^T (+pos, +mask) for two 16-key tiles ----
    f32x8 s[2];
#pragma unroll
    for (int st = 0; st < 2; ++st) {
      bf16x16 kb0 = ld_b_bf16(&ldsK[st * 16 + r][hi * 16]);
      bf16x16 kb1 = ld_b_bf16(&ldsK[st * 16 + r][32 + hi * 16]);
      f32x8 t = zero8();
      t = wmma_bf16(qa0, kb0, t);
      t = wmma_bf16(qa1, kb1, t);
      const int tg = tc + st * 16 + r;          // this lane's key column
      const bool msk = mrow[tg] != 0;
#pragma unroll
      for (int j = 0; j < 8; ++j) {
        const int qg = q0 + j + 8 * hi;
        int d = tg - qg;
        d = d < -Ln ? -Ln : (d > Ln ? Ln : d);
        const float pos = (float)relbase[(size_t)qg * RELW + (d + Ln)];
        t[j] = msk ? -1e30f : (t[j] + pos);
      }
      s[st] = t;
    }

    // ---- online softmax update ----
    float al[8];
#pragma unroll
    for (int j = 0; j < 8; ++j) {
      float cm = fmaxf(s[0][j], s[1][j]);
#pragma unroll
      for (int off = 1; off < 16; off <<= 1)
        cm = fmaxf(cm, __shfl_xor(cm, off, 32));
      const float mnew = fmaxf(mi[j], cm);
      al[j] = __expf(mi[j] - mnew);
      const float p0 = __expf(s[0][j] - mnew);
      const float p1 = __expf(s[1][j] - mnew);
      s[0][j] = p0; s[1][j] = p1;
      float sum = p0 + p1;
#pragma unroll
      for (int off = 1; off < 16; off <<= 1)
        sum += __shfl_xor(sum, off, 32);
      li[j] = li[j] * al[j] + sum;
      mi[j] = mnew;
    }
#pragma unroll
    for (int nt = 0; nt < 4; ++nt)
#pragma unroll
      for (int j = 0; j < 8; ++j) acc[nt][j] *= al[j];

    // ---- transpose P (C-layout -> A-layout) via per-wave LDS ----
#pragma unroll
    for (int j = 0; j < 8; ++j) {
      ldsP[wid][j + 8 * hi][r]      = (bf16)s[0][j];
      ldsP[wid][j + 8 * hi][16 + r] = (bf16)s[1][j];
    }
    __builtin_amdgcn_wave_barrier();
    asm volatile("s_wait_dscnt 0" ::: "memory");  // DS in-order per wave
    const bf16x16 pa = ld_a_bf16(&ldsP[wid][r][0], hi);

    // ---- O += P V ----
#pragma unroll
    for (int nt = 0; nt < 4; ++nt) {
      bf16x16 vb = ld_b_bf16(&ldsV[nt * 16 + r][hi * 16]);
      acc[nt] = wmma_bf16(pa, vb, acc[nt]);
    }
  }

  // ---- normalize and write heads-concat layout (B,T,D) ----
#pragma unroll
  for (int j = 0; j < 8; ++j) li[j] = 1.0f / li[j];
#pragma unroll
  for (int nt = 0; nt < 4; ++nt) {
#pragma unroll
    for (int j = 0; j < 8; ++j) {
      const int qg = q0 + j + 8 * hi;
      comb[((size_t)b * Tn + qg) * Dn + h * DH + nt * 16 + r] =
          (bf16)(acc[nt][j] * li[j]);
    }
  }
}

// ---------------------------------------------------------------------------
// Final GEMM: out = comb @ Wo^T.  comb bf16 (4096,1024), Wo bf16, out f32.
// ---------------------------------------------------------------------------
__global__ void __launch_bounds__(256) gemm_final_kernel(
    const bf16* __restrict__ A, const bf16* __restrict__ W,
    float* __restrict__ out) {
  const int lane = threadIdx.x & 31;
  const int r = lane & 15, hi = lane >> 4;
  const int wid = threadIdx.x >> 5;
  const int m0 = blockIdx.x * 64 + (wid & 3) * 16;
  const int n0 = blockIdx.y * 128 + (wid >> 2) * 64;

  f32x8 acc[4];
#pragma unroll
  for (int nt = 0; nt < 4; ++nt) acc[nt] = zero8();

  const bf16* arow = A + (size_t)(m0 + r) * Dn;
  const bf16* brow0 = W + (size_t)(n0 + r) * Dn;
  for (int k0 = 0; k0 < Dn; k0 += 32) {
    if (k0 + 256 < Dn) __builtin_prefetch(arow + k0 + 256, 0, 1);
    bf16x16 a = ld_a_bf16(arow + k0, hi);
    bf16x16 b0 = ld_b_bf16(brow0 + (size_t)0 * 16 * Dn + k0 + hi * 16);
    bf16x16 b1 = ld_b_bf16(brow0 + (size_t)1 * 16 * Dn + k0 + hi * 16);
    bf16x16 b2 = ld_b_bf16(brow0 + (size_t)2 * 16 * Dn + k0 + hi * 16);
    bf16x16 b3 = ld_b_bf16(brow0 + (size_t)3 * 16 * Dn + k0 + hi * 16);
    acc[0] = wmma_bf16(a, b0, acc[0]);
    acc[1] = wmma_bf16(a, b1, acc[1]);
    acc[2] = wmma_bf16(a, b2, acc[2]);
    acc[3] = wmma_bf16(a, b3, acc[3]);
  }
#pragma unroll
  for (int nt = 0; nt < 4; ++nt)
#pragma unroll
    for (int j = 0; j < 8; ++j)
      out[(size_t)(m0 + j + 8 * hi) * Dn + n0 + nt * 16 + r] = acc[nt][j];
}

// ---------------------------------------------------------------------------
extern "C" void kernel_launch(void* const* d_in, const int* in_sizes, int n_in,
                              void* d_out, int out_size, void* d_ws,
                              size_t ws_size, hipStream_t stream) {
  (void)in_sizes; (void)n_in; (void)out_size; (void)ws_size;
  const float* x_q = (const float*)d_in[0];
  const float* x_k = (const float*)d_in[1];
  const float* x_v = (const float*)d_in[2];
  const unsigned char* mask = (const unsigned char*)d_in[3];  // jnp bool: 1 byte
  const float* Wq = (const float*)d_in[4];
  const float* Wk = (const float*)d_in[5];
  const float* Wv = (const float*)d_in[6];
  const float* Wo = (const float*)d_in[7];
  const float* pos = (const float*)d_in[8];
  float* out = (float*)d_out;

  // workspace layout (all 16B aligned)
  char* ws = (char*)d_ws;
  const size_t szX   = (size_t)Bn * Tn * Dn * sizeof(bf16);       // 8 MB
  const size_t szW   = (size_t)Dn * Dn * sizeof(bf16);            // 2 MB
  const size_t szPos = ((size_t)(2 * Ln + 1) * DH * sizeof(bf16) + 4095) & ~4095ull;
  const size_t szQKV = (size_t)Bn * Hn * Tn * DH * sizeof(bf16);  // 8 MB
  const size_t szRel = (size_t)Bn * Hn * Tn * RELW * sizeof(bf16);

  size_t off = 0;
  bf16* Xqb = (bf16*)(ws + off); off += szX;
  bf16* Xkb = (bf16*)(ws + off); off += szX;
  bf16* Xvb = (bf16*)(ws + off); off += szX;
  bf16* Wqb = (bf16*)(ws + off); off += szW;
  bf16* Wkb = (bf16*)(ws + off); off += szW;
  bf16* Wvb = (bf16*)(ws + off); off += szW;
  bf16* Wob = (bf16*)(ws + off); off += szW;
  bf16* Pb  = (bf16*)(ws + off); off += szPos;
  bf16* Qbf = (bf16*)(ws + off); off += szQKV;
  bf16* Kbf = (bf16*)(ws + off); off += szQKV;
  bf16* Vt  = (bf16*)(ws + off); off += szQKV;
  bf16* rel = (bf16*)(ws + off); off += szRel;
  bf16* comb= (bf16*)(ws + off); off += szX;

  const dim3 blk(256);
  auto cvt_launch = [&](const float* s, bf16* d, size_t n) {
    const unsigned blocks = (unsigned)((n / 8 + 255) / 256);
    cvt_bf16_kernel<<<dim3(blocks), blk, 0, stream>>>(s, d, (unsigned)n);
  };
  cvt_launch(x_q, Xqb, (size_t)Bn * Tn * Dn);
  cvt_launch(x_k, Xkb, (size_t)Bn * Tn * Dn);
  cvt_launch(x_v, Xvb, (size_t)Bn * Tn * Dn);
  cvt_launch(Wq, Wqb, (size_t)Dn * Dn);
  cvt_launch(Wk, Wkb, (size_t)Dn * Dn);
  cvt_launch(Wv, Wvb, (size_t)Dn * Dn);
  cvt_launch(Wo, Wob, (size_t)Dn * Dn);
  cvt_launch(pos, Pb, (size_t)(2 * Ln + 1) * DH);

  const dim3 gproj(Bn * Tn / 64, Dn / 128);  // (64, 8)

  // Fold 1/SCALE (=1/8) into Q: scales both QK^T and Q.pos_emb terms.
  gemm_proj_kernel<<<gproj, blk, 0, stream>>>(Xqb, Wqb, Qbf, 0.125f, 0);
  gemm_proj_kernel<<<gproj, blk, 0, stream>>>(Xkb, Wkb, Kbf, 1.0f, 0);
  gemm_proj_kernel<<<gproj, blk, 0, stream>>>(Xvb, Wvb, Vt, 1.0f, 1);
  relscore_kernel<<<dim3(Bn * Hn * Tn / 128), blk, 0, stream>>>(Qbf, Pb, rel);
  attn_kernel<<<dim3(Tn / 128, Bn * Hn), blk, 0, stream>>>(Qbf, Kbf, Vt, rel,
                                                           mask, comb);
  gemm_final_kernel<<<gproj, blk, 0, stream>>>(comb, Wob, out);
}
